// RNN_SRU_74388833567160
// MI455X (gfx1250) — matrix operand order repaired and checked by reading, not compiled, not run on hardware
//
#include <hip/hip_runtime.h>
#include <hip/hip_bf16.h>

#define L_SEQ  512
#define BATCH  32
#define NIN    72
#define DHALF  512        // D
#define DFULL  1024       // 2D
#define NOUT   144
#define MROWS  (L_SEQ*BATCH)   // 16384

#define USE_TDM 1

typedef __attribute__((ext_vector_type(16))) __bf16        v16bf;
typedef __attribute__((ext_vector_type(8)))  float         v8f;
typedef __attribute__((ext_vector_type(4)))  unsigned int  u32x4;
typedef __attribute__((ext_vector_type(8)))  int           i32x8;
typedef __attribute__((ext_vector_type(4)))  int           i32x4;

union FragAB { v16bf v; u32x4 q[2]; };

// ---------------- TDM: async 2D bf16 tile (rows x 32) -> LDS with 16B row pad ----------------
// D# layout per CDNA5 ISA. data_size=2B, pad 64B->80B rows,
// which matches the LDS layout of 40 bf16 per row used by the WMMA fragment loads.
__device__ __forceinline__ void tdm_load_tile(const void* gaddr, unsigned lds_off,
                                              unsigned tensor_d0 /*K elems*/,
                                              unsigned tensor_d1 /*rows*/,
                                              unsigned tile_rows /*128*/) {
  unsigned long long ga = (unsigned long long)(uintptr_t)gaddr;
  u32x4 g0;
  g0.x = 1u;                                        // count=1, is_restore=0, gather off
  g0.y = lds_off;                                   // [63:32]  lds_addr (bytes)
  g0.z = (unsigned)(ga & 0xFFFFFFFFu);              // [95:64]  global_addr lo
  g0.w = (unsigned)((ga >> 32) & 0x01FFFFFFu)       // [120:96] global_addr hi
       | (2u << 30);                                // [127:126] type = 2 (image)
  const unsigned flags = (1u << 16)                 // data_size = 1 -> 2 bytes
                       | (1u << 20)                 // pad_enable
                       | (3u << 22)                 // pad_interval: 16 DWORDs (=64B)
                       | (3u << 25);                // pad_amount:   4 DWORDs (=16B)
  union { unsigned long long q[4]; i32x8 v; } g1;
  // bits [31:0] flags | [47:32] atomic_barrier_addr=0 | [79:48] tensor_dim0
  g1.q[0] = (unsigned long long)flags
          | (((unsigned long long)tensor_d0 & 0xFFFFull) << 48);
  // bits [79:64] tensor_dim0 hi | [111:80] tensor_dim1 | [127:112] tile_dim0
  g1.q[1] = ((unsigned long long)(tensor_d0 >> 16) & 0xFFFFull)
          | ((unsigned long long)tensor_d1 << 16)
          | (32ull << 48);                          // tile_dim0 = 32 elems (64B)
  // bits [143:128] tile_dim1 | [159:144] tile_dim2=0 | [207:160] tensor_dim0_stride lo32
  g1.q[2] = ((unsigned long long)tile_rows & 0xFFFFull)
          | (((unsigned long long)tensor_d0 & 0xFFFFFFFFull) << 32);
  // bits [207:192] tensor_dim0_stride hi16 | [255:224.. ] tensor_dim1_stride = 0
  g1.q[3] = ((unsigned long long)tensor_d0 >> 32) & 0xFFFFull;
  i32x4 z4 = {0, 0, 0, 0};
#if __clang_major__ >= 23
  i32x8 z8 = {0, 0, 0, 0, 0, 0, 0, 0};
  __builtin_amdgcn_tensor_load_to_lds(g0, g1.v, z4, z4, z8, 0);
#else
  __builtin_amdgcn_tensor_load_to_lds(g0, g1.v, z4, z4, 0);
#endif
}

// ---------- weight transpose: f32 (K,N) -> bf16 (N,K) ----------
__global__ void k_transpose_w(const float* __restrict__ W, unsigned short* __restrict__ Wt,
                              int K, int N) {
  __shared__ float tile[32][33];
  int nb = blockIdx.x * 32;
  int kb = blockIdx.y * 32;
  int tx = threadIdx.x, ty = threadIdx.y;            // 32 x 8 threads
  #pragma unroll
  for (int i = 0; i < 32; i += 8)
    tile[ty + i][tx] = W[(size_t)(kb + ty + i) * N + nb + tx];
  __syncthreads();
  #pragma unroll
  for (int i = 0; i < 32; i += 8)
    ((__hip_bfloat16*)Wt)[(size_t)(nb + ty + i) * K + kb + tx] =
        __float2bfloat16(tile[tx][ty + i]);
}

// ---------- fused lin1 + ReLU + LayerNorm -> bf16 row (width 512) ----------
__global__ void k_lin1_ln(const float* __restrict__ x, const float* __restrict__ w,
                          const float* __restrict__ bias, const float* __restrict__ g,
                          const float* __restrict__ beta, unsigned short* __restrict__ A) {
  int row = blockIdx.x;            // token index
  int tid = threadIdx.x;           // 128 threads
  __shared__ float xs[NIN];
  __shared__ float rs[128], rq[128];
  if (tid < NIN) xs[tid] = x[(size_t)row * NIN + tid];
  __syncthreads();
  float acc[4];
  #pragma unroll
  for (int j = 0; j < 4; ++j) acc[j] = bias[tid + j * 128];
  for (int kk = 0; kk < NIN; ++kk) {
    float xv = xs[kk];
    #pragma unroll
    for (int j = 0; j < 4; ++j)
      acc[j] = fmaf(xv, w[(size_t)kk * DHALF + tid + j * 128], acc[j]);
  }
  float s = 0.f, q = 0.f;
  #pragma unroll
  for (int j = 0; j < 4; ++j) {
    acc[j] = fmaxf(acc[j], 0.f);
    s += acc[j]; q += acc[j] * acc[j];
  }
  rs[tid] = s; rq[tid] = q;
  __syncthreads();
  for (int off = 64; off > 0; off >>= 1) {
    if (tid < off) { rs[tid] += rs[tid + off]; rq[tid] += rq[tid + off]; }
    __syncthreads();
  }
  float mean = rs[0] * (1.f / DHALF);
  float var  = rq[0] * (1.f / DHALF) - mean * mean;
  float inv  = rsqrtf(var + 1e-5f);
  #pragma unroll
  for (int j = 0; j < 4; ++j) {
    int d = tid + j * 128;
    ((__hip_bfloat16*)A)[(size_t)row * DHALF + d] =
        __float2bfloat16((acc[j] - mean) * inv * g[d] + beta[d]);
  }
}

// ---------- LayerNorm over width-1024 rows -> bf16 ----------
__global__ void k_layernorm(const float* __restrict__ H, const float* __restrict__ g,
                            const float* __restrict__ beta, unsigned short* __restrict__ A) {
  int row = blockIdx.x;
  int tid = threadIdx.x;           // 256 threads
  __shared__ float rs[256], rq[256];
  float v[4]; float s = 0.f, q = 0.f;
  #pragma unroll
  for (int j = 0; j < 4; ++j) {
    v[j] = H[(size_t)row * DFULL + tid + j * 256];
    s += v[j]; q += v[j] * v[j];
  }
  rs[tid] = s; rq[tid] = q;
  __syncthreads();
  for (int off = 128; off > 0; off >>= 1) {
    if (tid < off) { rs[tid] += rs[tid + off]; rq[tid] += rq[tid + off]; }
    __syncthreads();
  }
  float mean = rs[0] * (1.f / DFULL);
  float var  = rq[0] * (1.f / DFULL) - mean * mean;
  float inv  = rsqrtf(var + 1e-5f);
  #pragma unroll
  for (int j = 0; j < 4; ++j) {
    int c = tid + j * 256;
    ((__hip_bfloat16*)A)[(size_t)row * DFULL + c] =
        __float2bfloat16((v[j] - mean) * inv * g[c] + beta[c]);
  }
}

// ---------- bf16 WMMA GEMM: C(MxN,f32) = A(MxK,bf16 row-major) * Wt(NxK,bf16)^T ----------
// TDM double-buffered: wave 0 issues tensor_load_to_lds for the next K slab while all
// 8 waves run WMMAs on the current buffer; sync = s_wait_tensorcnt + workgroup barrier.
__global__ __launch_bounds__(256)
void k_gemm_bf16(const unsigned short* __restrict__ A, const unsigned short* __restrict__ Wt,
                 float* __restrict__ C, int M, int N, int K) {
  __shared__ __align__(16) unsigned short sA[2][128 * 40];  // 128 rows x 32, padded to 40
  __shared__ __align__(16) unsigned short sB[2][128 * 40];
  const int tid  = threadIdx.x;
  const int lane = tid & 31;
  const int wave = tid >> 5;
  const int wm   = (wave & 1) * 64;     // wave's M offset in tile
  const int wn   = (wave >> 1) * 32;    // wave's N offset in tile
  const int m0   = blockIdx.y * 128;
  const int n0   = blockIdx.x * 128;

  v8f acc[4][2];
  for (int mi = 0; mi < 4; ++mi)
    for (int ni = 0; ni < 2; ++ni)
      for (int e = 0; e < 8; ++e) acc[mi][ni][e] = 0.f;

  const int mrow = lane & 15;           // sub-tile row (A) / col (B)
  const int kbA  = (lane >> 4) * 8;     // A: half-wave K split 0..7 / 8..15 (+16 hi chunk)
  const int kbB  = (lane >> 4) * 8;     // B: half-wave K split 0..7 / 8..15 (+16 hi chunk)

#if USE_TDM
  if (wave == 0) {
    tdm_load_tile(&A [(size_t)m0 * K], (unsigned)(uintptr_t)&sA[0][0], K, M, 128);
    tdm_load_tile(&Wt[(size_t)n0 * K], (unsigned)(uintptr_t)&sB[0][0], K, N, 128);
    __builtin_amdgcn_s_wait_tensorcnt(0);
  }
  __syncthreads();
#endif

  int buf = 0;
  for (int k0 = 0; k0 < K; k0 += 32) {
#if USE_TDM
    const bool more = (k0 + 32) < K;
    if (wave == 0 && more) {            // kick off DMA of next slab into the other buffer
      tdm_load_tile(&A [(size_t)m0 * K + k0 + 32], (unsigned)(uintptr_t)&sA[buf ^ 1][0],
                    K, M, 128);
      tdm_load_tile(&Wt[(size_t)n0 * K + k0 + 32], (unsigned)(uintptr_t)&sB[buf ^ 1][0],
                    K, N, 128);
    }
#else
    {  // straight-line cooperative copy: 2 chunks/thread, all loads issued before stores
      int r0 = tid >> 2, qd = (tid & 3) * 8;
      u32x4 a0 = *(const u32x4*)&A [(size_t)(m0 + r0)      * K + k0 + qd];
      u32x4 a1 = *(const u32x4*)&A [(size_t)(m0 + r0 + 64) * K + k0 + qd];
      u32x4 b0 = *(const u32x4*)&Wt[(size_t)(n0 + r0)      * K + k0 + qd];
      u32x4 b1 = *(const u32x4*)&Wt[(size_t)(n0 + r0 + 64) * K + k0 + qd];
      *(u32x4*)&sA[0][ r0       * 40 + qd] = a0;
      *(u32x4*)&sA[0][(r0 + 64) * 40 + qd] = a1;
      *(u32x4*)&sB[0][ r0       * 40 + qd] = b0;
      *(u32x4*)&sB[0][(r0 + 64) * 40 + qd] = b1;
    }
    __syncthreads();
#endif

    FragAB fa[4], fb[2];
    #pragma unroll
    for (int mi = 0; mi < 4; ++mi) {
      int base = (wm + mi * 16 + mrow) * 40 + kbA;
      fa[mi].q[0] = *(const u32x4*)&sA[buf][base];        // K = kbA..kbA+7
      fa[mi].q[1] = *(const u32x4*)&sA[buf][base + 16];   // K = kbA+16..kbA+23
    }
    #pragma unroll
    for (int ni = 0; ni < 2; ++ni) {
      int base = (wn + ni * 16 + mrow) * 40 + kbB;
      fb[ni].q[0] = *(const u32x4*)&sB[buf][base];        // K = kbB..kbB+7
      fb[ni].q[1] = *(const u32x4*)&sB[buf][base + 16];   // K = kbB+16..kbB+23
    }
    #pragma unroll
    for (int mi = 0; mi < 4; ++mi)
      #pragma unroll
      for (int ni = 0; ni < 2; ++ni)
        acc[mi][ni] = __builtin_amdgcn_wmma_f32_16x16x32_bf16(
            false, fa[mi].v, false, fb[ni].v, (short)0, acc[mi][ni], false, false);

#if USE_TDM
    if (wave == 0 && more) __builtin_amdgcn_s_wait_tensorcnt(0);
    __syncthreads();
    buf ^= 1;
#else
    __syncthreads();
#endif
  }

  const int nCol = n0 + wn + (lane & 15);
  const int mTop = m0 + wm + ((lane >> 4) * 8);      // C layout: VGPR r -> row r / r+8
  #pragma unroll
  for (int mi = 0; mi < 4; ++mi)
    #pragma unroll
    for (int ni = 0; ni < 2; ++ni)
      #pragma unroll
      for (int r = 0; r < 8; ++r)
        C[(size_t)(mTop + mi * 16 + r) * N + nCol + ni * 16] = acc[mi][ni][r];
}

// ---------- bidirectional SRU scan: one thread per (dir, b, d) chain ----------
template <int KK>
__global__ void k_sru(const float* __restrict__ U, const unsigned short* __restrict__ Aln,
                      const float* __restrict__ wc, const float* __restrict__ bb,
                      float* __restrict__ H, float* __restrict__ cOut) {
  int idx = blockIdx.x * blockDim.x + threadIdx.x;   // 0..32767
  int d   = idx & (DHALF - 1);
  int b   = (idx >> 9) & (BATCH - 1);
  int dir = idx >> 14;
  const int N = DFULL * KK;                          // GEMM output width
  float wcf = wc[(0 * 2 + dir) * DHALF + d];
  float wcr = wc[(1 * 2 + dir) * DHALF + d];
  float bf  = bb[(0 * 2 + dir) * DHALF + d];
  float br  = bb[(1 * 2 + dir) * DHALF + d];
  float c = 0.f;
  const int colBase = dir * DHALF * KK + d * KK;
  const int hCol    = dir * DHALF + d;
  for (int t = 0; t < L_SEQ; ++t) {
    int tt = dir ? (L_SEQ - 1 - t) : t;              // backward dir runs reversed in time
    size_t rowb = (size_t)tt * BATCH + b;
    const float* u = U + rowb * N + colBase;
    float u0 = u[0], u1 = u[1], u2 = u[2];
    float res;
    if (KK == 4) res = u[3];
    else         res = __bfloat162float(((const __hip_bfloat16*)Aln)[rowb * DFULL + hCol]);
    float f = 1.f / (1.f + __expf(-(u1 + wcf * c + bf)));
    c = f * c + (1.f - f) * u0;
    float r = 1.f / (1.f + __expf(-(u2 + wcr * c + br)));
    H[rowb * DFULL + hCol] = r * c + (1.f - r) * res;
  }
  cOut[(size_t)b * DFULL + hCol] = c;                // final cell state
}

// ---------- final projection: out = H @ lin2_w + lin2_b ----------
__global__ void k_lin2(const float* __restrict__ H, const float* __restrict__ w,
                       const float* __restrict__ bias, float* __restrict__ out) {
  int row = blockIdx.x;
  int tid = threadIdx.x;                             // 160 threads (144 active)
  __shared__ float hrow[DFULL];
  for (int i = tid; i < DFULL; i += 160) hrow[i] = H[(size_t)row * DFULL + i];
  __syncthreads();
  if (tid < NOUT) {
    float a = bias[tid];
    for (int kk = 0; kk < DFULL; ++kk)
      a = fmaf(hrow[kk], w[(size_t)kk * NOUT + tid], a);
    out[(size_t)row * NOUT + tid] = a;
  }
}

extern "C" void kernel_launch(void* const* d_in, const int* in_sizes, int n_in,
                              void* d_out, int out_size, void* d_ws, size_t ws_size,
                              hipStream_t stream) {
  const float* x      = (const float*)d_in[0];
  const float* lin1_w = (const float*)d_in[1];
  const float* lin1_b = (const float*)d_in[2];
  const float* lin2_w = (const float*)d_in[3];
  const float* lin2_b = (const float*)d_in[4];
  const float* W0     = (const float*)d_in[5];
  const float* wc0    = (const float*)d_in[6];
  const float* b0     = (const float*)d_in[7];
  const float* ln0_g  = (const float*)d_in[8];
  const float* ln0_b  = (const float*)d_in[9];
  const float* Wl     = (const float*)d_in[10];
  const float* wcl    = (const float*)d_in[11];
  const float* bl     = (const float*)d_in[12];
  const float* lnl_g  = (const float*)d_in[13];
  const float* lnl_b  = (const float*)d_in[14];

  char* p = (char*)d_ws;
  unsigned short* Abf = (unsigned short*)p; p += (size_t)MROWS * DFULL * 2;   // 32 MB bf16 acts
  float*          U   = (float*)p;          p += (size_t)MROWS * 4096 * 4;    // 256 MB U (max)
  float*          H   = (float*)p;          p += (size_t)MROWS * DFULL * 4;   // 64 MB h
  unsigned short* Wt0 = (unsigned short*)p; p += (size_t)4096 * DHALF * 2;    // 4 MB
  unsigned short* WtL = (unsigned short*)p;                                   // 4 x 6 MB

  float* out = (float*)d_out;
  float* cs  = out + (size_t)MROWS * NOUT;  // c outputs appended after main output

  dim3 tb(32, 8);
  k_transpose_w<<<dim3(4096 / 32, 512 / 32), tb, 0, stream>>>(W0, Wt0, 512, 4096);
  for (int i = 0; i < 4; ++i)
    k_transpose_w<<<dim3(3072 / 32, 1024 / 32), tb, 0, stream>>>(
        Wl + (size_t)i * 1024 * 3072, WtL + (size_t)i * 3072 * 1024, 1024, 3072);

  // Layer 0: lin1+relu+LN -> bf16, GEMM (K=512, N=4096), scan k=4
  k_lin1_ln<<<MROWS, 128, 0, stream>>>(x, lin1_w, lin1_b, ln0_g, ln0_b, Abf);
  k_gemm_bf16<<<dim3(4096 / 128, MROWS / 128), 256, 0, stream>>>(Abf, Wt0, U, MROWS, 4096, 512);
  k_sru<4><<<128, 256, 0, stream>>>(U, Abf, wc0, b0, H, cs);

  // Layers 1..4: LN -> bf16, GEMM (K=1024, N=3072), scan k=3 (residual from LN output)
  for (int i = 0; i < 4; ++i) {
    k_layernorm<<<MROWS, 256, 0, stream>>>(H, lnl_g + (size_t)i * DFULL,
                                           lnl_b + (size_t)i * DFULL, Abf);
    k_gemm_bf16<<<dim3(3072 / 128, MROWS / 128), 256, 0, stream>>>(
        Abf, WtL + (size_t)i * 3072 * 1024, U, MROWS, 3072, 1024);
    k_sru<3><<<128, 256, 0, stream>>>(U, Abf, wcl + (size_t)i * 2048, bl + (size_t)i * 2048,
                                      H, cs + (size_t)(i + 1) * BATCH * DFULL);
  }

  k_lin2<<<MROWS, 160, 0, stream>>>(H, lin2_w, lin2_b, out);
}
